// SparseProtoLinear_89395449299717
// MI455X (gfx1250) — matrix operand order repaired and checked by reading, not compiled, not run on hardware
//
#include <hip/hip_runtime.h>
#include <cmath>

typedef __attribute__((ext_vector_type(16))) _Float16 v16h;
typedef __attribute__((ext_vector_type(8)))  _Float16 v8h;
typedef __attribute__((ext_vector_type(4)))  _Float16 v4h;
typedef __attribute__((ext_vector_type(8)))  float    v8f;

#define S_TOTAL   (4 * 2048 * 16)     // 131072 tokens
#define DH        64
#define P_EXP     8
#define NBLK      256
#define NTHR      128                 // 4 waves (wave32)
#define WAVES_PB  4
#define TILES     (S_TOTAL / 16)      // 8192
#define TOT_WAVES (NBLK * WAVES_PB)   // 1024
#define TILES_PW  (TILES / TOT_WAVES) // 8

// dynamic LDS layout (bytes)
#define OFF_W1H    0
#define OFF_W2H    (OFF_W1H + P_EXP * DH * DH * 2)    // 65536
#define OFF_PROTO  (OFF_W2H + P_EXP * DH * DH * 2)    // 131072
#define OFF_GATE   (OFF_PROTO + P_EXP * DH * 4)       // 133120
#define OFF_WAVE   (OFF_GATE + 32)                    // 133152
#define WAVE_BYTES 12800  // xf0 4096 | xf1 4096 | xh 2048 | hm 2048 | wt 512
#define LDS_BYTES  (OFF_WAVE + WAVES_PB * WAVE_BYTES) // 184352

__device__ __forceinline__ void lds_fence() {
    asm volatile("s_wait_dscnt 0" ::: "memory");
}
__device__ __forceinline__ void async_fence() {
    asm volatile("s_wait_asynccnt 0" ::: "memory");
}

// Async-stage one 16x64 f32 tile (4 KB) global -> LDS. Each b128 moves
// 32 lanes x 16B = 512B; INST_OFFSET walks both LDS and global sides.
__device__ __forceinline__ void async_stage_tile(const float* gbase, float* dst, int lane) {
    unsigned ldsaddr = (unsigned)(uintptr_t)dst + (unsigned)(lane * 16);
    unsigned voff    = (unsigned)(lane * 16);
    #pragma unroll
    for (int i = 0; i < 8; ++i) {
        asm volatile("global_load_async_to_lds_b128 %0, %1, %2 offset:%3"
                     :: "v"(ldsaddr), "v"(voff), "s"(gbase), "n"(i * 512)
                     : "memory");
    }
}

// A-fragment (16x32 f16, MxK): lane holds row m=lane&15; khalf=lane>>4 selects
// K sub-pattern {khalf*8 .. +7} and {16+khalf*8 .. +7} (ISA 7.12.2 table).
__device__ __forceinline__ v16h load_frag_A(const _Float16* rowbase, int khalf) {
    v8h lo = *reinterpret_cast<const v8h*>(rowbase + khalf * 8);
    v8h hi = *reinterpret_cast<const v8h*>(rowbase + khalf * 8 + 16);
    return __builtin_shufflevector(lo, hi, 0,1,2,3,4,5,6,7,8,9,10,11,12,13,14,15);
}

// B-fragment (32x16 f16, KxN): lane holds column n=lane&15; khalf selects
// K {khalf*8 .. +7} and {16+khalf*8 .. +7} (same order as A).
__device__ __forceinline__ v16h load_frag_B(const _Float16* colbase, int khalf) {
    v8h lo = *reinterpret_cast<const v8h*>(colbase + khalf * 8);
    v8h hi = *reinterpret_cast<const v8h*>(colbase + khalf * 8 + 16);
    return __builtin_shufflevector(lo, hi, 0,1,2,3,4,5,6,7,8,9,10,11,12,13,14,15);
}

__device__ __forceinline__ v8f wmma_f16(v16h a, v16h b, v8f c) {
    return __builtin_amdgcn_wmma_f32_16x16x32_f16(
        /*neg_a=*/false, a, /*neg_b=*/false, b,
        /*c_mod=*/(short)0, c, /*reuse_a=*/false, /*reuse_b=*/false);
}

extern "C" __global__ void __launch_bounds__(NTHR, 1)
spl_kernel(const float* __restrict__ x, const float* __restrict__ proto,
           const float* __restrict__ gate, const float* __restrict__ w1,
           const float* __restrict__ w2, float* __restrict__ out_y,
           float* __restrict__ out_logits, float* __restrict__ out_mask) {
    extern __shared__ __align__(16) char smem[];
    _Float16* w1h    = reinterpret_cast<_Float16*>(smem + OFF_W1H);
    _Float16* w2h    = reinterpret_cast<_Float16*>(smem + OFF_W2H);
    float*    protoL = reinterpret_cast<float*>(smem + OFF_PROTO);
    float*    gateL  = reinterpret_cast<float*>(smem + OFF_GATE);

    const int tid  = threadIdx.x;
    const int lane = tid & 31;
    const int wave = tid >> 5;

    char* my = smem + OFF_WAVE + wave * WAVE_BYTES;
    float*    xf0 = reinterpret_cast<float*>(my);             // [16][64] f32 buf A
    float*    xf1 = reinterpret_cast<float*>(my + 4096);      // [16][64] f32 buf B
    _Float16* xh  = reinterpret_cast<_Float16*>(my + 8192);   // [16][64] f16
    _Float16* hm  = reinterpret_cast<_Float16*>(my + 10240);  // [16][64] f16 (hmid)
    float*    wt  = reinterpret_cast<float*>(my + 12288);     // [16][8]  gate weights

    // ---- one-time: cache expert weights (f16) + proto/gate in LDS ----
    for (int i = tid; i < P_EXP * DH * DH; i += NTHR) {
        w1h[i] = (_Float16)w1[i];
        w2h[i] = (_Float16)w2[i];
    }
    for (int i = tid; i < P_EXP * DH; i += NTHR) protoL[i] = proto[i];
    if (tid < P_EXP) gateL[tid] = gate[tid];
    __syncthreads();

    const int gwave = blockIdx.x * WAVES_PB + wave;
    const int m     = lane & 15;   // token row within tile == N column for B frags
    const int khalf = lane >> 4;

    // prefetch first tile
    async_stage_tile(x + (size_t)gwave * 16 * DH, xf0, lane);

    for (int it = 0; it < TILES_PW; ++it) {
        const int tile = gwave + it * TOT_WAVES;
        const int s0   = tile * 16;
        float* xfc = (it & 1) ? xf1 : xf0;
        float* xfn = (it & 1) ? xf0 : xf1;

        async_fence();   // current tile resident in LDS
        if (it + 1 < TILES_PW) {   // uniform: prefetch next tile
            async_stage_tile(x + (size_t)(tile + TOT_WAVES) * 16 * DH, xfn, lane);
        }

        // ---- build f16 copy of x tile for WMMA A-fragments ----
        #pragma unroll
        for (int i = 0; i < 8; ++i) {
            const int idx = i * 128 + lane * 4;
            float4 v = *reinterpret_cast<const float4*>(xfc + idx);
            v4h h = { (_Float16)v.x, (_Float16)v.y, (_Float16)v.z, (_Float16)v.w };
            *reinterpret_cast<v4h*>(xh + idx) = h;
        }

        // ---- routing (fp32): lane -> token m, experts p0..p0+3 ----
        const int p0 = khalf * 4;
        float acc[4] = {0.f, 0.f, 0.f, 0.f};
        const float4* xr = reinterpret_cast<const float4*>(xfc + m * DH);
        const float4* pr = reinterpret_cast<const float4*>(protoL);
        #pragma unroll
        for (int d4 = 0; d4 < 16; ++d4) {
            float4 xv = xr[d4];
            #pragma unroll
            for (int i = 0; i < 4; ++i) {
                float4 q = pr[(p0 + i) * 16 + d4];
                acc[i] += xv.x * q.x + xv.y * q.y + xv.z * q.z + xv.w * q.w;
            }
        }
        float lg[4], mk[4], wv[4];
        #pragma unroll
        for (int i = 0; i < 4; ++i) {
            lg[i] = acc[i] * 0.125f - gateL[p0 + i];
            mk[i] = fmaxf(lg[i], 0.f);
            wv[i] = (mk[i] > 1e-6f) ? mk[i] : 0.f;
        }
        float4 lg4 = {lg[0], lg[1], lg[2], lg[3]};
        float4 mk4 = {mk[0], mk[1], mk[2], mk[3]};
        float4 wv4 = {wv[0], wv[1], wv[2], wv[3]};
        *reinterpret_cast<float4*>(out_logits + (size_t)(s0 + m) * P_EXP + p0) = lg4;
        *reinterpret_cast<float4*>(out_mask   + (size_t)(s0 + m) * P_EXP + p0) = mk4;
        *reinterpret_cast<float4*>(wt + m * P_EXP + p0) = wv4;
        lds_fence();

        // ---- x A-fragments (reused across all experts) ----
        v16h a1_0 = load_frag_A(xh + m * DH + 0,  khalf);
        v16h a1_1 = load_frag_A(xh + m * DH + 32, khalf);

        v8f yacc[4];
        #pragma unroll
        for (int db = 0; db < 4; ++db) { v8f z = {}; yacc[db] = z; }

        for (int p = 0; p < P_EXP; ++p) {
            const _Float16* w1p = w1h + p * DH * DH;
            const _Float16* w2p = w2h + p * DH * DH;

            // per-row gate weights: single base, constant strides -> DS imm offsets
            const float* wb = wt + (8 * khalf) * P_EXP + p;
            float wrow[8];
            #pragma unroll
            for (int r = 0; r < 8; ++r) wrow[r] = wb[r * P_EXP];

            // layer 1: hmid = silu(X @ W1^T), gate-weight folded in
            #pragma unroll
            for (int nb = 0; nb < 4; ++nb) {
                v8f h = {};
                v16h b0 = load_frag_B(w1p + (nb * 16 + m) * DH + 0,  khalf);
                v16h b1 = load_frag_B(w1p + (nb * 16 + m) * DH + 32, khalf);
                h = wmma_f16(a1_0, b0, h);
                h = wmma_f16(a1_1, b1, h);
                _Float16* hb = hm + (8 * khalf) * DH + nb * 16 + m;
                #pragma unroll
                for (int r = 0; r < 8; ++r) {
                    float v  = h[r];
                    // silu via fast rcp: v * 1/(1+exp(-v))
                    float sv = v * __builtin_amdgcn_rcpf(1.f + __expf(-v));
                    hb[r * DH] = (_Float16)(sv * wrow[r]);
                }
            }
            lds_fence();

            // layer 2: y += (wt .* silu(...)) @ W2^T  (accumulates across experts)
            v16h a2_0 = load_frag_A(hm + m * DH + 0,  khalf);
            v16h a2_1 = load_frag_A(hm + m * DH + 32, khalf);
            #pragma unroll
            for (int db = 0; db < 4; ++db) {
                v16h c0 = load_frag_B(w2p + (db * 16 + m) * DH + 0,  khalf);
                v16h c1 = load_frag_B(w2p + (db * 16 + m) * DH + 32, khalf);
                yacc[db] = wmma_f16(a2_0, c0, yacc[db]);
                yacc[db] = wmma_f16(a2_1, c1, yacc[db]);
            }
        }

        // ---- write y tile: lane l, VGPR r -> (row r+8*khalf, col db*16+n) ----
        float* yb = out_y + (size_t)(s0 + 8 * khalf) * DH + m;
        #pragma unroll
        for (int db = 0; db < 4; ++db) {
            #pragma unroll
            for (int r = 0; r < 8; ++r) {
                yb[(size_t)r * DH + db * 16] = yacc[db][r];
            }
        }
    }
}

extern "C" void kernel_launch(void* const* d_in, const int* in_sizes, int n_in,
                              void* d_out, int out_size, void* d_ws, size_t ws_size,
                              hipStream_t stream) {
    (void)in_sizes; (void)n_in; (void)out_size; (void)d_ws; (void)ws_size;
    const float* x     = (const float*)d_in[0];
    const float* proto = (const float*)d_in[1];
    const float* gate  = (const float*)d_in[2];
    const float* w1    = (const float*)d_in[3];
    const float* w2    = (const float*)d_in[4];

    float* out_y      = (float*)d_out;                               // [S,64]
    float* out_logits = out_y + (size_t)S_TOTAL * DH;                // [S,8]
    float* out_mask   = out_logits + (size_t)S_TOTAL * P_EXP;        // [S,8]

    hipFuncSetAttribute((const void*)spl_kernel,
                        hipFuncAttributeMaxDynamicSharedMemorySize, (int)LDS_BYTES);
    spl_kernel<<<NBLK, NTHR, LDS_BYTES, stream>>>(x, proto, gate, w1, w2,
                                                  out_y, out_logits, out_mask);
}
